// SchNetEncoder_protein_44848048505534
// MI455X (gfx1250) — matrix operand order repaired and checked by reading, not compiled, not run on hardware
//
#include <hip/hip_runtime.h>
#include <hip/hip_bf16.h>
#include <math.h>
#include <stdint.h>

// ---------------- CDNA5 WMMA types ----------------
typedef __attribute__((ext_vector_type(16))) __bf16 v16bf;
typedef __attribute__((ext_vector_type(8)))  float  v8f;

#define NNODE   50000
#define NEDGE   800000
#define INDIM   27
#define HDIM    128
#define GDIM    64
#define NBLK    6
#define CUT_R   10.0f
#define LOG2_C  0.6931471805599453f
#define PI_C    3.14159265358979323846f

// gaussian smearing constants: offset_k = k * (CUT/(G-1)), coeff = -0.5/step^2
#define GSTEP   (CUT_R / (float)(GDIM - 1))
#define GCOEF   (-0.5f / (GSTEP * GSTEP))

// shifted softplus
__device__ __forceinline__ float sspf(float x) {
  float sp = (x > 20.0f) ? x : log1pf(__expf(x));
  return sp - LOG2_C;
}

// A-matrix (16-bit, 16x32) element->K map: lane half selects K-subblock.
// e in [0,16): k = e + (e>=8 ? 8 : 0) + 8*half   (ISA 7.12.2)
__device__ __forceinline__ int kmapA(int e, int half) {
  return e + ((e >= 8) ? 8 : 0) + (half << 3);
}

// Load a pre-shuffled B fragment (16x32-per-K-tile weight blob, NT=8).
__device__ __forceinline__ v16bf load_bfrag(const __bf16* base, int kt, int nt, int lane) {
  return *(const v16bf*)(base + ((((kt << 3) + nt) << 5) + lane) * 16);
}

// bf16 (packed in u32, halfword sel) -> f32
__device__ __forceinline__ float bfbits_to_f(unsigned int u, int hi) {
  return __uint_as_float(hi ? (u & 0xffff0000u) : (u << 16));
}

// ---------------- K0: pack f32 weight [K,N] into fragment-ready bf16 blob ----------------
// blob element tid = ((kt*NT+nt)*32+lane)*16+e  holds  W[kt*32 + half*8 + e + (e>=8 ? 8 : 0)][nt*16 + (lane&15)]
__global__ void pack_bfrag_k(const float* __restrict__ w, __bf16* __restrict__ out,
                             int K, int N) {
  int tid = blockIdx.x * blockDim.x + threadIdx.x;
  int NT = N >> 4;
  int e = tid & 15;
  int lane = (tid >> 4) & 31;
  int rest = tid >> 9;
  int nt = rest % NT;
  int kt = rest / NT;
  int k = (kt << 5) + ((lane >> 4) << 3) + e + ((e >= 8) ? 8 : 0);
  int n = (nt << 4) + (lane & 15);
  out[tid] = (__bf16)w[k * N + n];
}

// ---------------- K1: edge geometry: d[e], cutoff gate C[e] ----------------
__global__ void edge_geom_k(const float* __restrict__ pos, const int* __restrict__ src,
                            const int* __restrict__ dst, float* __restrict__ dE,
                            float* __restrict__ Cg) {
  int e = blockIdx.x * blockDim.x + threadIdx.x;
  int s = src[e], t = dst[e];
  float dx = pos[s * 3 + 0] - pos[t * 3 + 0];
  float dy = pos[s * 3 + 1] - pos[t * 3 + 1];
  float dz = pos[s * 3 + 2] - pos[t * 3 + 2];
  float d = sqrtf(dx * dx + dy * dy + dz * dz);
  float C = 0.5f * (__cosf(d * (PI_C / CUT_R)) + 1.0f);
  C = (d <= CUT_R) ? C : 0.0f;
  dE[e] = d;
  Cg[e] = C;
}

// ---------------- K2: embedding h = node_attr @ emb_w + emb_b ----------------
__global__ void embed_k(const float* __restrict__ na, const float* __restrict__ w,
                        const float* __restrict__ b, float* __restrict__ h) {
  int tid = blockIdx.x * blockDim.x + threadIdx.x;
  int n = tid >> 7, c = tid & 127;
  float acc = b[c];
#pragma unroll
  for (int k = 0; k < INDIM; ++k) acc += na[n * INDIM + k] * w[k * HDIM + c];
  h[tid] = acc;
}

// ---------------- K3: xf = bf16(h @ lin1) ; one wave per 16-node tile ----------------
__global__ void __launch_bounds__(32)
xf_gemm_k(const float* __restrict__ h, const __bf16* __restrict__ wfrag,
          __bf16* __restrict__ xf) {
  int lane = threadIdx.x & 31;
  int half = lane >> 4, mrow = lane & 15, col = lane & 15;
  long base = (long)blockIdx.x * 16;

  v16bf a[4];
#pragma unroll
  for (int kt = 0; kt < 4; ++kt) {
#pragma unroll
    for (int e = 0; e < 16; ++e)
      a[kt][e] = (__bf16)h[(base + mrow) * HDIM + (kt << 5) + kmapA(e, half)];
  }
#pragma unroll
  for (int nt = 0; nt < 8; ++nt) {
    v8f acc = {};
#pragma unroll
    for (int kt = 0; kt < 4; ++kt) {
      v16bf bf = load_bfrag(wfrag, kt, nt, lane);
      acc = __builtin_amdgcn_wmma_f32_16x16x32_bf16(false, a[kt], false, bf,
                                                    (short)0, acc, false, false);
    }
#pragma unroll
    for (int i = 0; i < 8; ++i)
      xf[(base + i + half * 8) * HDIM + (nt << 4) + col] = (__bf16)acc[i];
  }
}

// ---------------- K4: fused edge pipeline ----------------
// async gather xf[src] -> LDS (overlapped) ; gaussians(d) -> WMMA nn1 -> +b -> ssp
// -> (LDS relayout) -> WMMA nn2 -> +b -> *C -> *xf -> atomicAdd agg[dst]
__global__ void __launch_bounds__(128)
edge_fused_k(const __bf16* __restrict__ nn1f, const float* __restrict__ nn1b,
             const __bf16* __restrict__ nn2f, const float* __restrict__ nn2b,
             const float* __restrict__ dE, const float* __restrict__ Cg,
             const __bf16* __restrict__ xf, const int* __restrict__ src,
             const int* __restrict__ dst, float* __restrict__ agg) {
  __shared__ unsigned int xf_lds[4][16 * 64];  // 16 rows x 128 bf16 (256B/row), per wave
  __shared__ float        t1[4][16 * HDIM];    // ssp intermediate, per wave

  int w = threadIdx.x >> 5;
  int lane = threadIdx.x & 31;
  int half = lane >> 4, mrow = lane & 15, col = lane & 15;
  long tile = (long)blockIdx.x * 4 + w;
  long e0 = tile * 16;

  // ---- ASYNC gather: 16 source rows (256B each) straight into LDS, no VGPR staging.
  // One global_load_async_to_lds_b128 moves 512B (2 rows) across the wave:
  // half-wave `sub` handles row 2i+sub, lane covers 16B chunk (lane&15)*16.
  {
    unsigned lds_base = (unsigned)(uintptr_t)&xf_lds[w][0];
    int sub = lane >> 4;
    int boff = (lane & 15) << 4;
#pragma unroll
    for (int i = 0; i < 8; ++i) {
      int m = (i << 1) + sub;
      long row = src[e0 + m];
      const char* gaddr = (const char*)xf + row * (HDIM * 2) + boff;
      unsigned laddr = lds_base + m * (HDIM * 2) + boff;
      asm volatile("global_load_async_to_lds_b128 %0, %1, off"
                   :: "v"(laddr), "v"(gaddr)
                   : "memory");
    }
  }

  // ---- A fragments: gaussian smearing of this lane's edge distance (K=64 -> 2 tiles)
  float dv = dE[e0 + mrow];
  v16bf ag[2];
#pragma unroll
  for (int kt = 0; kt < 2; ++kt) {
#pragma unroll
    for (int e = 0; e < 16; ++e) {
      int k = (kt << 5) + kmapA(e, half);
      float t = dv - (float)k * GSTEP;
      ag[kt][e] = (__bf16)__expf(GCOEF * t * t);
    }
  }

  // ---- GEMM1: [16x64] x [64x128] -> ssp -> t1 (LDS); async gather still in flight
#pragma unroll
  for (int nt = 0; nt < 8; ++nt) {
    v8f acc = {};
#pragma unroll
    for (int kt = 0; kt < 2; ++kt) {
      v16bf bf = load_bfrag(nn1f, kt, nt, lane);
      acc = __builtin_amdgcn_wmma_f32_16x16x32_bf16(false, ag[kt], false, bf,
                                                    (short)0, acc, false, false);
    }
    float bias = nn1b[(nt << 4) + col];
#pragma unroll
    for (int i = 0; i < 8; ++i)
      t1[w][(i + half * 8) * HDIM + (nt << 4) + col] = sspf(acc[i] + bias);
  }

  // drain the async gather, then make both LDS surfaces visible
  asm volatile("s_wait_asynccnt 0" ::: "memory");
  __syncthreads();

  // ---- A fragments for GEMM2 from LDS (K=128 -> 4 tiles)
  v16bf at[4];
#pragma unroll
  for (int kt = 0; kt < 4; ++kt) {
#pragma unroll
    for (int e = 0; e < 16; ++e)
      at[kt][e] = (__bf16)t1[w][mrow * HDIM + (kt << 5) + kmapA(e, half)];
  }

  // per-D-row edge metadata (rows this lane owns: m = i + 8*half)
  int   drow[8];
  float crow[8];
#pragma unroll
  for (int i = 0; i < 8; ++i) {
    int m = i + half * 8;
    drow[i] = dst[e0 + m];
    crow[i] = Cg[e0 + m];
  }

  // ---- GEMM2: [16x128] x [128x128]; gate, gather-multiply, scatter-add
#pragma unroll
  for (int nt = 0; nt < 8; ++nt) {
    v8f acc = {};
#pragma unroll
    for (int kt = 0; kt < 4; ++kt) {
      v16bf bf = load_bfrag(nn2f, kt, nt, lane);
      acc = __builtin_amdgcn_wmma_f32_16x16x32_bf16(false, at[kt], false, bf,
                                                    (short)0, acc, false, false);
    }
    int n = (nt << 4) + col;
    float bias = nn2b[n];
#pragma unroll
    for (int i = 0; i < 8; ++i) {
      int m = i + half * 8;
      unsigned int xu = xf_lds[w][m * 64 + (n >> 1)];
      float xv = bfbits_to_f(xu, n & 1);
      float msg = (acc[i] + bias) * crow[i] * xv;
      atomicAdd(&agg[(long)drow[i] * HDIM + n], msg);
    }
  }
}

// ---------------- K5: node update: h += ssp(agg@lin2 + b2) @ lin + b ----------------
__global__ void __launch_bounds__(32)
node_update_k(const float* __restrict__ agg, const __bf16* __restrict__ lin2f,
              const float* __restrict__ lin2b, const __bf16* __restrict__ linf,
              const float* __restrict__ linb, float* __restrict__ h) {
  __shared__ float st[16 * HDIM];
  int lane = threadIdx.x & 31;
  int half = lane >> 4, mrow = lane & 15, col = lane & 15;
  long base = (long)blockIdx.x * 16;

  v16bf a[4];
#pragma unroll
  for (int kt = 0; kt < 4; ++kt) {
#pragma unroll
    for (int e = 0; e < 16; ++e)
      a[kt][e] = (__bf16)agg[(base + mrow) * HDIM + (kt << 5) + kmapA(e, half)];
  }
#pragma unroll
  for (int nt = 0; nt < 8; ++nt) {
    v8f acc = {};
#pragma unroll
    for (int kt = 0; kt < 4; ++kt) {
      v16bf bf = load_bfrag(lin2f, kt, nt, lane);
      acc = __builtin_amdgcn_wmma_f32_16x16x32_bf16(false, a[kt], false, bf,
                                                    (short)0, acc, false, false);
    }
    float bias = lin2b[(nt << 4) + col];
#pragma unroll
    for (int i = 0; i < 8; ++i)
      st[(i + half * 8) * HDIM + (nt << 4) + col] = sspf(acc[i] + bias);
  }
  __syncthreads();

  v16bf t[4];
#pragma unroll
  for (int kt = 0; kt < 4; ++kt) {
#pragma unroll
    for (int e = 0; e < 16; ++e)
      t[kt][e] = (__bf16)st[mrow * HDIM + (kt << 5) + kmapA(e, half)];
  }
#pragma unroll
  for (int nt = 0; nt < 8; ++nt) {
    v8f acc = {};
#pragma unroll
    for (int kt = 0; kt < 4; ++kt) {
      v16bf bf = load_bfrag(linf, kt, nt, lane);
      acc = __builtin_amdgcn_wmma_f32_16x16x32_bf16(false, t[kt], false, bf,
                                                    (short)0, acc, false, false);
    }
    float bias = linb[(nt << 4) + col];
#pragma unroll
    for (int i = 0; i < 8; ++i) {
      long idx = (base + i + half * 8) * HDIM + (nt << 4) + col;
      h[idx] += acc[i] + bias;
    }
  }
}

// ---------------- host driver ----------------
extern "C" void kernel_launch(void* const* d_in, const int* in_sizes, int n_in,
                              void* d_out, int out_size, void* d_ws, size_t ws_size,
                              hipStream_t stream) {
  const float* node_attr = (const float*)d_in[0];
  const float* pos       = (const float*)d_in[1];
  // d_in[2] = batch (unused: single graph)
  const int*   edge_index = (const int*)d_in[3];
  const float* emb_w  = (const float*)d_in[4];
  const float* emb_b  = (const float*)d_in[5];
  const float* lin1_w = (const float*)d_in[6];
  const float* nn1_w  = (const float*)d_in[7];
  const float* nn1_b  = (const float*)d_in[8];
  const float* nn2_w  = (const float*)d_in[9];
  const float* nn2_b  = (const float*)d_in[10];
  const float* lin2_w = (const float*)d_in[11];
  const float* lin2_b = (const float*)d_in[12];
  const float* lin_w  = (const float*)d_in[13];
  const float* lin_b  = (const float*)d_in[14];

  const int* src = edge_index;
  const int* dst = edge_index + NEDGE;
  float* h = (float*)d_out;  // [N,H] lives in the output buffer

  // workspace carve-up (~46 MB total)
  char* ws = (char*)d_ws;
  size_t off = 0;
  auto carve = [&](size_t bytes) {
    size_t o = off;
    off = (off + bytes + 255) & ~(size_t)255;
    return o;
  };
  __bf16* nn1f  = (__bf16*)(ws + carve((size_t)NBLK * GDIM * HDIM * 2));
  __bf16* nn2f  = (__bf16*)(ws + carve((size_t)NBLK * HDIM * HDIM * 2));
  __bf16* lin1f = (__bf16*)(ws + carve((size_t)NBLK * HDIM * HDIM * 2));
  __bf16* lin2f = (__bf16*)(ws + carve((size_t)NBLK * HDIM * HDIM * 2));
  __bf16* linf  = (__bf16*)(ws + carve((size_t)NBLK * HDIM * HDIM * 2));
  float*  dE    = (float*)(ws + carve((size_t)NEDGE * 4));
  float*  Cg    = (float*)(ws + carve((size_t)NEDGE * 4));
  __bf16* xf    = (__bf16*)(ws + carve((size_t)NNODE * HDIM * 2));
  float*  agg   = (float*)(ws + carve((size_t)NNODE * HDIM * 4));

  // pack weights into fragment-ready bf16 blobs
  for (int b = 0; b < NBLK; ++b) {
    pack_bfrag_k<<<(GDIM * HDIM) / 256, 256, 0, stream>>>(
        nn1_w + (size_t)b * GDIM * HDIM, nn1f + (size_t)b * GDIM * HDIM, GDIM, HDIM);
    pack_bfrag_k<<<(HDIM * HDIM) / 256, 256, 0, stream>>>(
        nn2_w + (size_t)b * HDIM * HDIM, nn2f + (size_t)b * HDIM * HDIM, HDIM, HDIM);
    pack_bfrag_k<<<(HDIM * HDIM) / 256, 256, 0, stream>>>(
        lin1_w + (size_t)b * HDIM * HDIM, lin1f + (size_t)b * HDIM * HDIM, HDIM, HDIM);
    pack_bfrag_k<<<(HDIM * HDIM) / 256, 256, 0, stream>>>(
        lin2_w + (size_t)b * HDIM * HDIM, lin2f + (size_t)b * HDIM * HDIM, HDIM, HDIM);
    pack_bfrag_k<<<(HDIM * HDIM) / 256, 256, 0, stream>>>(
        lin_w + (size_t)b * HDIM * HDIM, linf + (size_t)b * HDIM * HDIM, HDIM, HDIM);
  }

  edge_geom_k<<<NEDGE / 256, 256, 0, stream>>>(pos, src, dst, dE, Cg);
  embed_k<<<(NNODE * HDIM) / 256, 256, 0, stream>>>(node_attr, emb_w, emb_b, h);

  for (int b = 0; b < NBLK; ++b) {
    xf_gemm_k<<<NNODE / 16, 32, 0, stream>>>(h, lin1f + (size_t)b * HDIM * HDIM, xf);
    hipMemsetAsync(agg, 0, (size_t)NNODE * HDIM * 4, stream);
    edge_fused_k<<<NEDGE / 64, 128, 0, stream>>>(
        nn1f + (size_t)b * GDIM * HDIM, nn1_b + (size_t)b * HDIM,
        nn2f + (size_t)b * HDIM * HDIM, nn2_b + (size_t)b * HDIM,
        dE, Cg, xf, src, dst, agg);
    node_update_k<<<NNODE / 16, 32, 0, stream>>>(
        agg, lin2f + (size_t)b * HDIM * HDIM, lin2_b + (size_t)b * HDIM,
        linf + (size_t)b * HDIM * HDIM, lin_b + (size_t)b * HDIM, h);
  }
}
